// MultiHeadSelfAttention_16466904613770
// MI455X (gfx1250) — matrix operand order repaired and checked by reading, not compiled, not run on hardware
//
#include <hip/hip_runtime.h>
#include <hip/hip_bf16.h>

typedef __bf16 bf16_t;
typedef __attribute__((ext_vector_type(16))) __bf16 v16bf;
typedef __attribute__((ext_vector_type(8)))  __bf16 v8bf;
typedef __attribute__((ext_vector_type(8)))  float  v8f;

#if defined(__gfx1250__) && __has_builtin(__builtin_amdgcn_global_load_async_to_lds_b128)
#define HAVE_ASYNC_LDS 1
#else
#define HAVE_ASYNC_LDS 0
#endif

#if HAVE_ASYNC_LDS
// Exact parameter types per hipcc diagnostic: <4 x i32> with AS1 (printed
// "__device__") for the global source; AS3 (LDS) for the destination.
typedef int v4i_ __attribute__((vector_size(16)));
typedef __attribute__((address_space(1))) v4i_ as1_v4i;
typedef __attribute__((address_space(3))) v4i_ as3_v4i;
#endif

// 16-byte global -> LDS copy. Async (ASYNCcnt-tracked, no VGPR round trip) when
// the gfx1250 async builtins are available; plain load+store otherwise.
static __device__ inline void cp16_g2l(const bf16_t* g, bf16_t* l) {
#if HAVE_ASYNC_LDS
    __builtin_amdgcn_global_load_async_to_lds_b128(
        (as1_v4i*)(void*)const_cast<bf16_t*>(g),
        (as3_v4i*)(void*)l,
        0, 0);
#else
    *(v8bf*)l = *(const v8bf*)g;
#endif
}

static __device__ inline void async_wait0() {
#if HAVE_ASYNC_LDS
#if __has_builtin(__builtin_amdgcn_s_wait_asynccnt)
    __builtin_amdgcn_s_wait_asynccnt(0);
#else
    asm volatile("s_wait_asynccnt 0x0" ::: "memory");
#endif
#endif
}

static __device__ inline v8f vzero8() {
    v8f v;
#pragma unroll
    for (int i = 0; i < 8; ++i) v[i] = 0.0f;
    return v;
}

// ---------------------------------------------------------------------------
// WMMA fragment loaders, per CDNA5 ISA 7.12.2 (wave32):
//  A 16x32 bf16 : lane L -> row m = L&15, kb = 8*(L>=16);
//                 elems 0..7 = k(kb..kb+7), elems 8..15 = k(kb+16..kb+23)
//  B 32x16 bf16 : lane L -> col n = L&15, kb = 8*(L>=16);
//                 elems 0..7 = k(kb..kb+7), elems 8..15 = k(kb+16..kb+23)
//  C 16x16 f32  : VGPR r -> row r + 8*(L>=16), col n = L&15
// ---------------------------------------------------------------------------
static __device__ inline v16bf frag_a_ld(const bf16_t* p0, int stride, int lane) {
    const int m  = lane & 15;
    const int kb = (lane >> 4) << 3;
    const bf16_t* p = p0 + m * stride + kb;
    union { v16bf v; v8bf h[2]; } u;
    u.h[0] = *(const v8bf*)(p);
    u.h[1] = *(const v8bf*)(p + 16);
    return u.v;
}

static __device__ inline v16bf frag_b_ld(const bf16_t* p0, int stride, int lane) {
    const int n  = lane & 15;
    const int kb = (lane >> 4) << 3;
    const bf16_t* p = p0 + n * stride + kb;
    union { v16bf v; v8bf h[2]; } u;
    u.h[0] = *(const v8bf*)(p);
    u.h[1] = *(const v8bf*)(p + 16);
    return u.v;
}

// ---------------------------------------------------------------------------
// f32 -> bf16 conversion
// ---------------------------------------------------------------------------
__global__ __launch_bounds__(256) void cvt_f32_bf16(const float* __restrict__ in,
                                                    bf16_t* __restrict__ out,
                                                    size_t n) {
    size_t i = (size_t)blockIdx.x * blockDim.x + threadIdx.x;
    const size_t stride = (size_t)gridDim.x * blockDim.x;
    for (; i < n; i += stride) out[i] = (bf16_t)in[i];
}

// ---------------------------------------------------------------------------
// GEMM: C[M,N] = A[M,K] * W[N,K]^T   (A,W bf16, accumulate f32)
// Block tile 128x128, BK=32, double-buffered LDS with async global->LDS
// staging. 256 threads = 8 waves, each wave does 32x64 (2x4 wmma tiles).
// ---------------------------------------------------------------------------
template <bool OUT_BF16>
__global__ __launch_bounds__(256) void gemm_xt_bf16(const bf16_t* __restrict__ A,
                                                    const bf16_t* __restrict__ W,
                                                    void* __restrict__ C,
                                                    int M, int N, int K) {
    __shared__ bf16_t sA[2][128 * 40];   // +8 bf16 pad keeps 16B alignment
    __shared__ bf16_t sB[2][128 * 40];

    const int tid  = threadIdx.x;
    const int lane = tid & 31;
    const int wid  = tid >> 5;
    const int wm   = (wid & 3) << 5;   // 0,32,64,96
    const int wn   = (wid >> 2) << 6;  // 0,64
    const size_t bm0 = (size_t)blockIdx.y << 7;
    const size_t bn0 = (size_t)blockIdx.x << 7;

    auto issue_tile = [&](int k0, int buf) {
#pragma unroll
        for (int t = 0; t < 2; ++t) {
            const int idx = tid + (t << 8);        // 0..511
            const int r   = idx >> 2;              // 0..127
            const int c8  = (idx & 3) << 3;        // 0,8,16,24
            cp16_g2l(&A[(bm0 + r) * K + k0 + c8], &sA[buf][r * 40 + c8]);
            cp16_g2l(&W[(bn0 + r) * K + k0 + c8], &sB[buf][r * 40 + c8]);
        }
    };

    v8f acc[2][4];
#pragma unroll
    for (int i = 0; i < 2; ++i)
#pragma unroll
        for (int j = 0; j < 4; ++j) acc[i][j] = vzero8();

    const int nk = K >> 5;
    issue_tile(0, 0);

    for (int kk = 0; kk < nk; ++kk) {
        const int buf = kk & 1;
        async_wait0();
        __syncthreads();
        if (kk + 1 < nk) issue_tile((kk + 1) << 5, buf ^ 1);  // overlaps compute below

        v16bf af[2], bfr[4];
#pragma unroll
        for (int mt = 0; mt < 2; ++mt) af[mt]  = frag_a_ld(&sA[buf][(wm + mt * 16) * 40], 40, lane);
#pragma unroll
        for (int nt = 0; nt < 4; ++nt) bfr[nt] = frag_b_ld(&sB[buf][(wn + nt * 16) * 40], 40, lane);
#pragma unroll
        for (int mt = 0; mt < 2; ++mt)
#pragma unroll
            for (int nt = 0; nt < 4; ++nt)
                acc[mt][nt] = __builtin_amdgcn_wmma_f32_16x16x32_bf16(
                    false, af[mt], false, bfr[nt], (short)0, acc[mt][nt], false, false);
    }

    const int n  = lane & 15;
    const int rb = (lane >> 4) << 3;
#pragma unroll
    for (int mt = 0; mt < 2; ++mt)
#pragma unroll
        for (int nt = 0; nt < 4; ++nt)
#pragma unroll
            for (int r = 0; r < 8; ++r) {
                const size_t gm = bm0 + wm + mt * 16 + rb + r;
                const size_t gn = bn0 + wn + nt * 16 + n;
                const float v = acc[mt][nt][r];
                if (OUT_BF16) ((bf16_t*)C)[gm * N + gn] = (bf16_t)v;
                else          ((float*)C)[gm * N + gn]  = v;
            }
}

// ---------------------------------------------------------------------------
// RoPE (in place, on Q and K slices of the packed qkv buffer)
// qkv layout: ((b*S + s)*3 + part)*2048 + h*128 + c
// ---------------------------------------------------------------------------
__global__ __launch_bounds__(256) void rope_qk(bf16_t* __restrict__ qkv,
                                               int B_, int S, int H) {
    const size_t total = (size_t)B_ * S * 2 * H * 64;
    size_t idx = (size_t)blockIdx.x * blockDim.x + threadIdx.x;
    if (idx >= total) return;
    const int i = (int)(idx & 63); idx >>= 6;
    const int h = (int)(idx % H);  idx /= H;
    const int part = (int)(idx & 1); idx >>= 1;
    const int s = (int)(idx % S);  idx /= S;
    const int b = (int)idx;

    // inv_freq = 10000^(-i/64); ln(10000) = 9.2103403719...
    const float inv = __expf(-(float)i * (9.210340371976184f / 64.0f));
    float sn, cs;
    __sincosf((float)s * inv, &sn, &cs);

    const size_t base = ((size_t)(b * S + s) * 3 + part) * 2048 + (size_t)h * 128 + 2 * i;
    const float x0 = (float)qkv[base];
    const float x1 = (float)qkv[base + 1];
    qkv[base]     = (bf16_t)(x0 * cs - x1 * sn);
    qkv[base + 1] = (bf16_t)(x0 * sn + x1 * cs);
}

// ---------------------------------------------------------------------------
// Flash attention (causal). 128 threads = 4 waves; each wave owns a 16-row
// query tile (block covers 64 query rows) of one (b,h). d_k = 128.
// K tile staged with async global->LDS; V staged transposed (manual scatter).
// ---------------------------------------------------------------------------
__global__ __launch_bounds__(128) void attn_fwd(const bf16_t* __restrict__ qkv,
                                                bf16_t* __restrict__ out,
                                                int B_, int S, int H) {
    __shared__ bf16_t sK[32 * 136];        // [key][dim], +8 pad
    __shared__ bf16_t sVt[128 * 40];       // [dim][key], +8 pad (transposed V)
    __shared__ bf16_t sP[4 * 16 * 40];     // per-wave P strip [row][key]

    const int tid  = threadIdx.x;
    const int lane = tid & 31;
    const int w    = tid >> 5;
    const int qb   = blockIdx.x;
    const int h    = blockIdx.y;
    const int b    = blockIdx.z;
    const int q0   = qb * 64 + w * 16;

    const size_t rs = (size_t)3 * 2048;    // row stride per sequence position
    const bf16_t* qbase = qkv + (size_t)b * S * rs + (size_t)h * 128;
    const bf16_t* kbase = qbase + 2048;
    const bf16_t* vbase = qbase + 4096;

    const int m  = lane & 15;
    const int hi = lane >> 4;

    // Q fragments: 4 x (16x32) covering d_k = 128, straight from global
    v16bf qf[4];
    {
        const bf16_t* qrow = qbase + (size_t)(q0 + m) * rs + (hi << 3);
#pragma unroll
        for (int kk = 0; kk < 4; ++kk) {
            union { v16bf v; v8bf h2[2]; } u;
            u.h2[0] = *(const v8bf*)(qrow + kk * 32);
            u.h2[1] = *(const v8bf*)(qrow + kk * 32 + 16);
            qf[kk] = u.v;
        }
    }

    v8f o[8];
#pragma unroll
    for (int t = 0; t < 8; ++t) o[t] = vzero8();
    float mrun[8], lrun[8];
#pragma unroll
    for (int r = 0; r < 8; ++r) { mrun[r] = -1e30f; lrun[r] = 0.0f; }

    const float scl = 0.08838834764831845f;   // 1/sqrt(128)
    const int nkb = (qb + 1) * 2;             // 32-key blocks up to block diagonal

    for (int kb = 0; kb < nkb; ++kb) {
        const int key0 = kb * 32;
        __syncthreads();
        // K: async copy; V: load + transposed scatter into sVt
#pragma unroll
        for (int t = 0; t < 4; ++t) {
            const int idx = tid + t * 128;      // 0..511
            const int row = idx >> 4;           // key 0..31
            const int c8  = (idx & 15) << 3;    // dim 0..120
            cp16_g2l(kbase + (size_t)(key0 + row) * rs + c8, &sK[row * 136 + c8]);
            const v8bf vv = *(const v8bf*)(vbase + (size_t)(key0 + row) * rs + c8);
#pragma unroll
            for (int j = 0; j < 8; ++j) sVt[(c8 + j) * 40 + row] = vv[j];
        }
        async_wait0();
        __syncthreads();

        // scores S = Q K^T for 16x32 keys -> two 16x16 C tiles
        v8f s[2];
        s[0] = vzero8(); s[1] = vzero8();
#pragma unroll
        for (int nt = 0; nt < 2; ++nt)
#pragma unroll
            for (int kk = 0; kk < 4; ++kk) {
                const v16bf kf = frag_b_ld(&sK[(nt * 16) * 136 + kk * 32], 136, lane);
                s[nt] = __builtin_amdgcn_wmma_f32_16x16x32_bf16(
                    false, qf[kk], false, kf, (short)0, s[nt], false, false);
            }

        // scale + causal mask in C-layout (lane holds col n, rows rb+0..7)
        const int nloc = lane & 15;
#pragma unroll
        for (int nt = 0; nt < 2; ++nt)
#pragma unroll
            for (int r = 0; r < 8; ++r) {
                const int keyg = key0 + nt * 16 + nloc;
                const int rowg = q0 + r + (hi << 3);
                const float v = s[nt][r] * scl;
                s[nt][r] = (keyg <= rowg) ? v : -1e30f;
            }

        // online softmax: row stats via 16-lane-half xor reductions
        float scale8[8];
#pragma unroll
        for (int r = 0; r < 8; ++r) {
            float v = fmaxf(s[0][r], s[1][r]);
#pragma unroll
            for (int off = 1; off < 16; off <<= 1) v = fmaxf(v, __shfl_xor(v, off, 32));
            const float mn = fmaxf(mrun[r], v);
            const float sc = __expf(mrun[r] - mn);
            const float p0 = (s[0][r] > -1e29f) ? __expf(s[0][r] - mn) : 0.0f;
            const float p1 = (s[1][r] > -1e29f) ? __expf(s[1][r] - mn) : 0.0f;
            float rsum = p0 + p1;
#pragma unroll
            for (int off = 1; off < 16; off <<= 1) rsum += __shfl_xor(rsum, off, 32);
            lrun[r] = lrun[r] * sc + rsum;
            mrun[r] = mn;
            scale8[r] = sc;
            s[0][r] = p0; s[1][r] = p1;
        }

        // restage P through LDS: C-layout -> A-layout
        bf16_t* sp = &sP[w * 16 * 40];
#pragma unroll
        for (int nt = 0; nt < 2; ++nt)
#pragma unroll
            for (int r = 0; r < 8; ++r)
                sp[(r + (hi << 3)) * 40 + nt * 16 + nloc] = (bf16_t)s[nt][r];
        __syncthreads();   // orders the wave-local DS store->load (uniform control flow)

        const v16bf pf = frag_a_ld(sp, 40, lane);

        // O = O*diag(scale) + P·V
#pragma unroll
        for (int nt2 = 0; nt2 < 8; ++nt2) {
#pragma unroll
            for (int r = 0; r < 8; ++r) o[nt2][r] *= scale8[r];
            const v16bf vf = frag_b_ld(&sVt[(nt2 * 16) * 40], 40, lane);
            o[nt2] = __builtin_amdgcn_wmma_f32_16x16x32_bf16(
                false, pf, false, vf, (short)0, o[nt2], false, false);
        }
    }

    // epilogue: O / l, store (B,S,H*dk) bf16
    float inv[8];
#pragma unroll
    for (int r = 0; r < 8; ++r) inv[r] = 1.0f / lrun[r];
    const int nloc = lane & 15;
#pragma unroll
    for (int nt2 = 0; nt2 < 8; ++nt2)
#pragma unroll
        for (int r = 0; r < 8; ++r) {
            const int rowg = q0 + r + (hi << 3);
            const int colg = h * 128 + nt2 * 16 + nloc;
            out[((size_t)b * S + rowg) * 2048 + colg] = (bf16_t)(o[nt2][r] * inv[r]);
        }
}

// ---------------------------------------------------------------------------
// Launch
// ---------------------------------------------------------------------------
extern "C" void kernel_launch(void* const* d_in, const int* in_sizes, int n_in,
                              void* d_out, int out_size, void* d_ws, size_t ws_size,
                              hipStream_t stream) {
    (void)in_sizes; (void)n_in; (void)out_size; (void)ws_size;
    const int B = 4, S = 2048, D = 2048, H = 16;

    const float* x     = (const float*)d_in[0];
    const float* w_qkv = (const float*)d_in[1];
    const float* w_out = (const float*)d_in[2];

    char* ws = (char*)d_ws;
    bf16_t* xb   = (bf16_t*)(ws + 0);            //  33.6 MB : x in bf16
    bf16_t* wqb  = (bf16_t*)(ws + 33554432);     //  25.2 MB : w_qkv bf16
    bf16_t* wob  = (bf16_t*)(ws + 58720256);     //   8.4 MB : w_out bf16
    bf16_t* qkv  = (bf16_t*)(ws + 67108864);     // 100.7 MB : QKV bf16 (RoPE in place)
    bf16_t* attn = (bf16_t*)(ws + 167772160);    //  33.6 MB : attention output bf16

    cvt_f32_bf16<<<4096, 256, 0, stream>>>(x, xb, (size_t)B * S * D);
    cvt_f32_bf16<<<4096, 256, 0, stream>>>(w_qkv, wqb, (size_t)3 * D * D);
    cvt_f32_bf16<<<2048, 256, 0, stream>>>(w_out, wob, (size_t)D * D);

    dim3 g1(3 * D / 128, B * S / 128);   // (48, 64)
    gemm_xt_bf16<true><<<g1, 256, 0, stream>>>(xb, wqb, (void*)qkv, B * S, 3 * D, D);

    const size_t nrope = (size_t)B * S * 2 * H * 64;
    rope_qk<<<(unsigned)((nrope + 255) / 256), 256, 0, stream>>>(qkv, B, S, H);

    dim3 g2(S / 64, H, B);               // (32, 16, 4)
    attn_fwd<<<g2, 128, 0, stream>>>(qkv, attn, B, S, H);

    dim3 g3(D / 128, B * S / 128);       // (16, 64)
    gemm_xt_bf16<false><<<g3, 256, 0, stream>>>(attn, wob, d_out, B * S, D, D);
}
